// RLAAttention_34789235098056
// MI455X (gfx1250) — matrix operand order repaired and checked by reading, not compiled, not run on hardware
//
#include <hip/hip_runtime.h>

// ---------------------------------------------------------------------------
// RLA (sketched) attention for MI455X / gfx1250, wave32 + WMMA bf16 + TDM.
//   B=2, S=2048, DIM=2048, 16 heads x 128 hd, 8 KV heads, SKETCH=64.
// All matmuls via v_wmma_f32_16x16x32_bf16 (fp32 accumulate).
// Attention is flash-style (online softmax) to avoid the 0.5 GB fp32 score tensor.
// K=64 GEMMs stage their shared 16-col B panel in LDS via the Tensor Data
// Mover (tensor_load_to_lds + s_wait_tensorcnt), cutting B traffic 4x/block.
// ---------------------------------------------------------------------------

typedef __attribute__((ext_vector_type(16))) __bf16 v16bf;
typedef __attribute__((ext_vector_type(8)))  float  v8f;
typedef __attribute__((ext_vector_type(4)))  unsigned int v4u;
typedef __attribute__((ext_vector_type(8)))  int    v8i;
typedef __attribute__((ext_vector_type(4)))  int    v4i;
typedef unsigned short u16;
typedef long long ll;

constexpr int kB   = 2;
constexpr int kS   = 2048;
constexpr int kD   = 2048;
constexpr int kNH  = 16;
constexpr int kHD  = 128;     // kD / kNH
constexpr int kKV  = 1024;    // 8 KV heads * 128
constexpr int kSK  = 64;
constexpr float kAttnScale = 0.088388347648318447f;   // 1/sqrt(128)
constexpr float kNegInf = -3.0e38f;

#if defined(__gfx1250__)
#  if __has_builtin(__builtin_amdgcn_tensor_load_to_lds) && \
      __has_builtin(__builtin_amdgcn_s_wait_tensorcnt)
#    define RLA_HAVE_TDM 1
#  else
#    define RLA_HAVE_TDM 0
#  endif
#else
#  define RLA_HAVE_TDM 0
#endif

__device__ __forceinline__ __bf16 f2bf(float f) {
  unsigned u = __builtin_bit_cast(unsigned, f);
  u16 s = (u16)((u + 0x7FFFu + ((u >> 16) & 1u)) >> 16);   // RNE
  return __builtin_bit_cast(__bf16, s);
}
__device__ __forceinline__ __bf16 u2bf(u16 s) { return __builtin_bit_cast(__bf16, s); }
__device__ __forceinline__ u16 bf2u(__bf16 b) { return __builtin_bit_cast(u16, b); }

// ---------------------------------------------------------------------------
// Generic batched WMMA GEMM: C = scale * op(A) @ op(B)
//   op(A): M x K   (TA: A stored K x M, i.e. read A[k*lda+m])
//   op(B): K x N   (TB: B stored N x K, i.e. read B[n*ldb+k])
//   A/B fp32 or bf16(u16) per template flags; C fp32 or bf16.
// One wave -> one 16x16 C tile; 4 waves / block; K stepped by 32.
// Fragment layouts used by this code (wave32; A/B k order is contiguous):
//   A lane L: row M = L&15; elem i -> K = i + 16*(L>>4)
//   B lane L: col N = L&15; elem i -> K = i + 16*(L>>4)
//   C lane L: col N = L&15; elem i -> row M = i + 8*(L>>4)
// STAGE_B: (requires !TA && !TB && BBF && K<=64 && exact grid) the block's
// shared KxN=Kx16 bf16 B panel is DMA'd to LDS once by the TDM, then all 4
// waves consume it from LDS while streaming their private A rows from HBM.
// ---------------------------------------------------------------------------
template<bool TA, bool TB, bool ABF, bool BBF, bool OBF, bool STAGE_B = false>
__global__ __launch_bounds__(128)
void wmma_gemm(const void* __restrict__ Ap, const void* __restrict__ Bp,
               void* __restrict__ Cp, int M, int N, int K,
               int lda, int ldb, int ldc, float scale,
               ll sA, ll sB, ll sC)
{
  const int wave = threadIdx.x >> 5;
  const int lane = threadIdx.x & 31;
  const int h = lane >> 4, r = lane & 15;
  const ll  bz = blockIdx.z;
  const int n0 = blockIdx.x * 16;
  const int m0 = (blockIdx.y * 4 + wave) * 16;

  if constexpr (STAGE_B) {
    __shared__ u16 ldsB[kSK * 16];             // K<=64 rows x 16 cols, 2 KB
    const ll bbase = bz * sB;
#if RLA_HAVE_TDM
    if (threadIdx.x == 0) {
      // --- Tensor DMA descriptor (D#), 2-D tile: 16 (contig) x K rows ---
      const unsigned long long gaddr =
          (unsigned long long)(const char*)Bp + (unsigned long long)((bbase + n0) * 2);
      const unsigned ldsaddr = (unsigned)(size_t)&ldsB[0];
      v4u g0;
      g0[0] = 1u;                                        // count=1, user D#
      g0[1] = ldsaddr;                                   // lds_addr
      g0[2] = (unsigned)gaddr;                           // global_addr[31:0]
      g0[3] = (unsigned)((gaddr >> 32) & 0x01FFFFFFull)  // global_addr[56:32]
              | (2u << 30);                              // type=2 (image)
      const unsigned td0 = (unsigned)ldb;                // tensor_dim0
      const unsigned td1 = (unsigned)K;                  // tensor_dim1
      const unsigned long long st0 = (unsigned long long)(unsigned)ldb;
      v8i g1;
      g1[0] = (int)(1u << 16);                           // data_size=1 (2 B)
      g1[1] = (int)((td0 & 0xFFFFu) << 16);              // dim0 @ bits79:48
      g1[2] = (int)((td0 >> 16) | ((td1 & 0xFFFFu) << 16));
      g1[3] = (int)((td1 >> 16) | (16u << 16));          // tile_dim0=16
      g1[4] = (int)(unsigned)K;                          // tile_dim1=K
      g1[5] = (int)(unsigned)(st0 & 0xFFFFFFFFull);      // dim0_stride lo
      g1[6] = (int)(unsigned)((st0 >> 32) & 0xFFFFull);  // dim0_stride hi
      g1[7] = 0;
      v4i gz = {0, 0, 0, 0};
#if __clang_major__ >= 23
      v8i gz8 = {0, 0, 0, 0, 0, 0, 0, 0};
      __builtin_amdgcn_tensor_load_to_lds(g0, g1, gz, gz, gz8, 0);
#else
      __builtin_amdgcn_tensor_load_to_lds(g0, g1, gz, gz, 0);
#endif
      __builtin_amdgcn_s_wait_tensorcnt(0);
    }
    __syncthreads();
#else
    for (int t = threadIdx.x; t < K * 16; t += 128) {
      const int kk = t >> 4, nn = t & 15;
      ldsB[t] = ((const u16*)Bp)[bbase + (ll)kk * ldb + n0 + nn];
    }
    __syncthreads();
#endif
    v8f acc = {};
    for (int k0 = 0; k0 < K; k0 += 32) {
      v16bf a, b;
#pragma unroll
      for (int i = 0; i < 16; ++i) {
        const int ka = k0 + i + 16 * h;
        const ll idxA = (ll)(m0 + r) * lda + ka + bz * sA;
        a[i] = ABF ? u2bf(((const u16*)Ap)[idxA]) : f2bf(((const float*)Ap)[idxA]);
        b[i] = u2bf(ldsB[(k0 + i + 16 * h) * 16 + r]);
      }
      acc = __builtin_amdgcn_wmma_f32_16x16x32_bf16(false, a, false, b,
                                                    (short)0, acc, false, false);
    }
#pragma unroll
    for (int i = 0; i < 8; ++i) {
      const int row = m0 + i + 8 * h;
      const ll idxC = (ll)row * ldc + (n0 + r) + bz * sC;
      const float v = acc[i] * scale;
      if (OBF) ((u16*)Cp)[idxC] = bf2u(f2bf(v));
      else     ((float*)Cp)[idxC] = v;
    }
    return;
  }

  if (m0 >= M || n0 >= N) return;          // whole-wave uniform exit

  v8f acc = {};
  for (int k0 = 0; k0 < K; k0 += 32) {
    v16bf a, b;
#pragma unroll
    for (int i = 0; i < 16; ++i) {
      const int ka = k0 + i + 16 * h;
      ll idxA = (TA ? (ll)ka * lda + (m0 + r) : (ll)(m0 + r) * lda + ka) + bz * sA;
      a[i] = ABF ? u2bf(((const u16*)Ap)[idxA]) : f2bf(((const float*)Ap)[idxA]);
      const int kb = k0 + i + 16 * h;
      ll idxB = (TB ? (ll)(n0 + r) * ldb + kb : (ll)kb * ldb + (n0 + r)) + bz * sB;
      b[i] = BBF ? u2bf(((const u16*)Bp)[idxB]) : f2bf(((const float*)Bp)[idxB]);
    }
    if (k0 + 32 < K) {   // speculative prefetch of next K tile -> global_prefetch_b8
      ll pA = (TA ? (ll)(k0 + 32) * lda + (m0 + r) : (ll)(m0 + r) * lda + (k0 + 32)) + bz * sA;
      __builtin_prefetch((const char*)Ap + pA * (ABF ? 2 : 4), 0, 0);
      ll pB = (TB ? (ll)(n0 + r) * ldb + (k0 + 32) : (ll)(k0 + 32) * ldb + (n0 + r)) + bz * sB;
      __builtin_prefetch((const char*)Bp + pB * (BBF ? 2 : 4), 0, 0);
    }
    acc = __builtin_amdgcn_wmma_f32_16x16x32_bf16(false, a, false, b,
                                                  (short)0, acc, false, false);
  }
#pragma unroll
  for (int i = 0; i < 8; ++i) {
    const int row = m0 + i + 8 * h;
    const ll idxC = (ll)row * ldc + (n0 + r) + bz * sC;
    const float v = acc[i] * scale;
    if (OBF) ((u16*)Cp)[idxC] = bf2u(f2bf(v));
    else     ((float*)Cp)[idxC] = v;
  }
}

// ---------------------------------------------------------------------------
// Rotary (faithful: out[2p] = x0*sin - x1*cos ; out[2p+1] = x0*cos + x1*sin)
// Reads fp32 (b,s,row_elems), writes bf16 bits, same layout.
// ---------------------------------------------------------------------------
__global__ void rope_pack(const float* __restrict__ in, const float* __restrict__ fc,
                          u16* __restrict__ out, int total_pairs, int row_elems)
{
  const int idx = blockIdx.x * blockDim.x + threadIdx.x;
  if (idx >= total_pairs) return;
  const int ppr = row_elems >> 1;
  const int pir = idx % ppr;
  const ll  bs  = idx / ppr;
  const int s   = (int)(bs % kS);
  const int hp  = pir % (kHD / 2);
  const float c  = fc[((ll)s * (kHD / 2) + hp) * 2 + 0];
  const float sn = fc[((ll)s * (kHD / 2) + hp) * 2 + 1];
  const float x0 = in[2 * (ll)idx];
  const float x1 = in[2 * (ll)idx + 1];
  out[2 * (ll)idx]     = bf2u(f2bf(x0 * sn - x1 * c));
  out[2 * (ll)idx + 1] = bf2u(f2bf(x0 * c + x1 * sn));
}

// ---------------------------------------------------------------------------
// Flash attention (causal, GQA n_rep=2). One wave per 16-row Q tile.
// Q: (b,s,DIM) bf16 ; K,V: (b,s,KV) bf16 ; Out: (b,s,DIM) fp32.
// 32 keys / iteration: 8 QK^T WMMAs + 8 PV WMMAs. P is re-laid out C->A
// through wave-private LDS (LDS ops are in-order within a wave on CDNA5).
// ---------------------------------------------------------------------------
__global__ __launch_bounds__(128)
void flash_attn(const u16* __restrict__ Qb, const u16* __restrict__ Kb,
                const u16* __restrict__ Vb, float* __restrict__ Out)
{
  __shared__ u16 ldsP[4][16][34];
  const int wave = threadIdx.x >> 5;
  const int lane = threadIdx.x & 31;
  const int hh = lane >> 4, r = lane & 15;
  const int q0  = (blockIdx.x * 4 + wave) * 16;
  const int hq  = blockIdx.y;
  const int kvh = hq >> 1;
  const int bB  = blockIdx.z;

  v16bf qa[4];
#pragma unroll
  for (int kc = 0; kc < 4; ++kc) {
#pragma unroll
    for (int i = 0; i < 16; ++i) {
      const int hd = kc * 32 + i + 16 * hh;
      qa[kc][i] = u2bf(Qb[((ll)bB * kS + q0 + r) * kD + hq * kHD + hd]);
    }
  }

  float mrow[8], lrow[8];
  v8f o[8] = {};
#pragma unroll
  for (int i = 0; i < 8; ++i) { mrow[i] = kNegInf; lrow[i] = 0.0f; }

  for (int kb0 = 0; kb0 <= q0 + 15; kb0 += 32) {
    v8f s0 = {}, s1 = {};
#pragma unroll
    for (int kc = 0; kc < 4; ++kc) {
      v16bf kf0, kf1;
#pragma unroll
      for (int i = 0; i < 16; ++i) {
        const int hd = kc * 32 + i + 16 * hh;
        int k0 = kb0 + r;      k0 = k0 < kS ? k0 : kS - 1;
        int k1 = kb0 + 16 + r; k1 = k1 < kS ? k1 : kS - 1;
        kf0[i] = u2bf(Kb[((ll)bB * kS + k0) * kKV + kvh * kHD + hd]);
        kf1[i] = u2bf(Kb[((ll)bB * kS + k1) * kKV + kvh * kHD + hd]);
      }
      s0 = __builtin_amdgcn_wmma_f32_16x16x32_bf16(false, qa[kc], false, kf0,
                                                   (short)0, s0, false, false);
      s1 = __builtin_amdgcn_wmma_f32_16x16x32_bf16(false, qa[kc], false, kf1,
                                                   (short)0, s1, false, false);
    }
    float corr[8];
#pragma unroll
    for (int i = 0; i < 8; ++i) {
      const int row = q0 + i + 8 * hh;
      float e0 = (kb0 + r      <= row) ? s0[i] * kAttnScale : kNegInf;
      float e1 = (kb0 + 16 + r <= row) ? s1[i] * kAttnScale : kNegInf;
      float t = fmaxf(e0, e1);
#pragma unroll
      for (int m = 1; m < 16; m <<= 1) t = fmaxf(t, __shfl_xor(t, m, 32));
      const float nm = fmaxf(mrow[i], t);
      const float c  = __expf(mrow[i] - nm);
      const float p0 = __expf(e0 - nm);
      const float p1 = __expf(e1 - nm);
      float sm = p0 + p1;
#pragma unroll
      for (int m = 1; m < 16; m <<= 1) sm += __shfl_xor(sm, m, 32);
      lrow[i] = lrow[i] * c + sm;
      mrow[i] = nm;
      corr[i] = c;
      ldsP[wave][i + 8 * hh][r]      = bf2u(f2bf(p0));
      ldsP[wave][i + 8 * hh][16 + r] = bf2u(f2bf(p1));
    }
#pragma unroll
    for (int j = 0; j < 8; ++j)
#pragma unroll
      for (int i = 0; i < 8; ++i) o[j][i] *= corr[i];

    v16bf pa;
#pragma unroll
    for (int i = 0; i < 16; ++i) {
      const int kcol = i + 16 * hh;
      pa[i] = u2bf(ldsP[wave][r][kcol]);
    }
#pragma unroll
    for (int j = 0; j < 8; ++j) {
      v16bf vf;
#pragma unroll
      for (int i = 0; i < 16; ++i) {
        int key = kb0 + i + 16 * hh; key = key < kS ? key : kS - 1;
        vf[i] = u2bf(Vb[((ll)bB * kS + key) * kKV + kvh * kHD + j * 16 + r]);
      }
      o[j] = __builtin_amdgcn_wmma_f32_16x16x32_bf16(false, pa, false, vf,
                                                     (short)0, o[j], false, false);
    }
  }

#pragma unroll
  for (int i = 0; i < 8; ++i) {
    const float inv = 1.0f / lrow[i];
    const int row = q0 + i + 8 * hh;
#pragma unroll
    for (int j = 0; j < 8; ++j)
      Out[((ll)bB * kS + row) * kD + hq * kHD + j * 16 + r] = o[j][i] * inv;
  }
}

// ---------------------------------------------------------------------------
extern "C" void kernel_launch(void* const* d_in, const int* in_sizes, int n_in,
                              void* d_out, int out_size, void* d_ws, size_t ws_size,
                              hipStream_t stream) {
  const float* x  = (const float*)d_in[0];
  const float* wq = (const float*)d_in[1];
  const float* wk = (const float*)d_in[2];
  const float* wv = (const float*)d_in[3];
  const float* wo = (const float*)d_in[4];
  const float* Sq = (const float*)d_in[5];
  const float* Sk = (const float*)d_in[6];
  const float* Sv = (const float*)d_in[7];
  const float* So = (const float*)d_in[8];
  const float* fc = (const float*)d_in[9];
  float* out = (float*)d_out;

  char* ws = (char*)d_ws;
  size_t off = 0;
  auto alloc = [&](size_t bytes) -> void* {
    void* p = ws + off;
    off = (off + bytes + 255) & ~(size_t)255;
    return p;
  };
  float* Qf  = (float*)alloc((size_t)kB * kS * kD  * 4);
  float* Kf  = (float*)alloc((size_t)kB * kS * kKV * 4);
  u16*   Qb  = (u16*)  alloc((size_t)kB * kS * kD  * 2);
  u16*   Kb  = (u16*)  alloc((size_t)kB * kS * kKV * 2);
  u16*   Vb  = (u16*)  alloc((size_t)kB * kS * kKV * 2);
  u16*   ASq = (u16*)  alloc((size_t)kB * kS * kSK * 2);
  u16*   ASk = (u16*)  alloc((size_t)kB * kS * kSK * 2);
  u16*   ASv = (u16*)  alloc((size_t)kB * kS * kSK * 2);
  u16*   ASo = (u16*)  alloc((size_t)kB * kS * kSK * 2);
  u16*   SBq = (u16*)  alloc((size_t)kB * kSK * kD  * 2);
  u16*   SBk = (u16*)  alloc((size_t)kB * kSK * kKV * 2);
  u16*   SBv = (u16*)  alloc((size_t)kB * kSK * kKV * 2);
  u16*   SBo = (u16*)  alloc((size_t)kB * kSK * kD  * 2);
  (void)ws_size; (void)in_sizes; (void)n_in; (void)out_size;

  const dim3 blk(128);
  const float invSK = 1.0f / (float)kSK;

  // --- stage 1: AS = x @ S / 64   (M=S, N=64, K=DIM), fp32 in -> bf16 out
  wmma_gemm<false,false,false,false,true><<<dim3(kSK/16, kS/64, kB), blk, 0, stream>>>(
      x, Sq, ASq, kS, kSK, kD, kD, kSK, kSK, invSK, (ll)kS*kD, (ll)kD*kSK, (ll)kS*kSK);
  wmma_gemm<false,false,false,false,true><<<dim3(kSK/16, kS/64, kB), blk, 0, stream>>>(
      x, Sk, ASk, kS, kSK, kD, kD, kSK, kSK, invSK, (ll)kS*kD, (ll)kD*kSK, (ll)kS*kSK);
  wmma_gemm<false,false,false,false,true><<<dim3(kSK/16, kS/64, kB), blk, 0, stream>>>(
      x, Sv, ASv, kS, kSK, kD, kD, kSK, kSK, invSK, (ll)kS*kD, (ll)kD*kSK, (ll)kS*kSK);

  // --- stage 2: SB = S^T @ W^T   (M=64, N=DIM|KV, K=DIM); A=S (KxM), B=W (NxK)
  wmma_gemm<true,true,false,false,true><<<dim3(kD/16, 1, kB), blk, 0, stream>>>(
      Sq, wq, SBq, kSK, kD, kD, kSK, kD, kD, 1.0f, (ll)kD*kSK, 0, (ll)kSK*kD);
  wmma_gemm<true,true,false,false,true><<<dim3(kKV/16, 1, kB), blk, 0, stream>>>(
      Sk, wk, SBk, kSK, kKV, kD, kSK, kD, kKV, 1.0f, (ll)kD*kSK, 0, (ll)kSK*kKV);
  wmma_gemm<true,true,false,false,true><<<dim3(kKV/16, 1, kB), blk, 0, stream>>>(
      Sv, wv, SBv, kSK, kKV, kD, kSK, kD, kKV, 1.0f, (ll)kD*kSK, 0, (ll)kSK*kKV);
  wmma_gemm<true,true,false,false,true><<<dim3(kD/16, 1, kB), blk, 0, stream>>>(
      So, wo, SBo, kSK, kD, kD, kSK, kD, kD, 1.0f, (ll)kD*kSK, 0, (ll)kSK*kD);

  // --- stage 3: Q/K fp32 (for RoPE), V straight to bf16 (K=64, TDM-staged B)
  wmma_gemm<false,false,true,true,false,true><<<dim3(kD/16, kS/64, kB), blk, 0, stream>>>(
      ASq, SBq, Qf, kS, kD, kSK, kSK, kD, kD, 1.0f, (ll)kS*kSK, (ll)kSK*kD, (ll)kS*kD);
  wmma_gemm<false,false,true,true,false,true><<<dim3(kKV/16, kS/64, kB), blk, 0, stream>>>(
      ASk, SBk, Kf, kS, kKV, kSK, kSK, kKV, kKV, 1.0f, (ll)kS*kSK, (ll)kSK*kKV, (ll)kS*kKV);
  wmma_gemm<false,false,true,true,true,true><<<dim3(kKV/16, kS/64, kB), blk, 0, stream>>>(
      ASv, SBv, Vb, kS, kKV, kSK, kSK, kKV, kKV, 1.0f, (ll)kS*kSK, (ll)kSK*kKV, (ll)kS*kKV);

  // --- stage 4: RoPE + bf16 pack
  {
    const int pq = kB * kS * kD / 2;
    rope_pack<<<(pq + 255) / 256, 256, 0, stream>>>(Qf, fc, Qb, pq, kD);
    const int pk = kB * kS * kKV / 2;
    rope_pack<<<(pk + 255) / 256, 256, 0, stream>>>(Kf, fc, Kb, pk, kKV);
  }

  // --- stage 5: causal flash attention -> d_out used as fp32 staging
  flash_attn<<<dim3(kS/64, kNH, kB), blk, 0, stream>>>(Qb, Kb, Vb, out);

  // --- stage 6: output sketch: ASo = attn_out @ So / 64 ; out = ASo @ SBo
  wmma_gemm<false,false,false,false,true><<<dim3(kSK/16, kS/64, kB), blk, 0, stream>>>(
      out, So, ASo, kS, kSK, kD, kD, kSK, kSK, invSK, (ll)kS*kD, (ll)kD*kSK, (ll)kS*kSK);
  wmma_gemm<false,false,true,true,false,true><<<dim3(kD/16, kS/64, kB), blk, 0, stream>>>(
      ASo, SBo, out, kS, kD, kSK, kSK, kD, kD, 1.0f, (ll)kS*kSK, (ll)kSK*kD, (ll)kS*kD);
}
